// StructAttnBlock_25598005084739
// MI455X (gfx1250) — matrix operand order repaired and checked by reading, not compiled, not run on hardware
//
#include <hip/hip_runtime.h>
#include <hip/hip_bf16.h>

// ---------------------------------------------------------------------------
// StructAttnBlock for MI455X (gfx1250)
//   bf16 WMMA pipeline with double-buffered GLOBAL_LOAD_ASYNC_TO_LDS staging.
//   B=4, N=2048, IN_DIM=HID=256, H=8, DH=32.
// ---------------------------------------------------------------------------

typedef __attribute__((ext_vector_type(16))) __bf16 v16bf;
typedef __attribute__((ext_vector_type(8)))  float  v8f;

__device__ __forceinline__ unsigned short f2bfu(float f) {
    unsigned int u = __builtin_bit_cast(unsigned int, f);
    u += 0x7FFFu + ((u >> 16) & 1u);        // round-to-nearest-even
    return (unsigned short)(u >> 16);
}
__device__ __forceinline__ __bf16 u2bf(unsigned short u) {
    return __builtin_bit_cast(__bf16, u);
}

__device__ __forceinline__ v8f wmma_bf16(v16bf a, v16bf b, v8f c) {
    return __builtin_amdgcn_wmma_f32_16x16x32_bf16(
        false, a, false, b, (short)0, c, false, false);
}

// --- CDNA5 async global->LDS copy (16B per lane), tracked by ASYNCcnt ------
__device__ __forceinline__ void async_copy_b128(void* lds, const void* g) {
    unsigned loff = (unsigned)(unsigned long long)lds;   // low 32b = LDS offset
    unsigned long long gaddr = (unsigned long long)g;
    asm volatile("global_load_async_to_lds_b128 %0, %1, off"
                 :: "v"(loff), "v"(gaddr)
                 : "memory");
}
template <int C>
__device__ __forceinline__ void wait_async() {
    asm volatile("s_wait_asynccnt %0" :: "i"(C) : "memory");
}

// ===========================================================================
// Kernel 0a: feats fp32 -> bf16 (packed stores).  2,097,152 elements.
// ===========================================================================
__global__ __launch_bounds__(256) void convert_feats_kernel(
    const float* __restrict__ in, unsigned int* __restrict__ out)
{
    int idx = (blockIdx.x * 256 + threadIdx.x) * 8;     // 8 floats / thread
    const float4* p = (const float4*)(in + idx);
    float4 a = p[0], b = p[1];
    uint4 w;
    w.x = ((unsigned)f2bfu(a.y) << 16) | f2bfu(a.x);
    w.y = ((unsigned)f2bfu(a.w) << 16) | f2bfu(a.z);
    w.z = ((unsigned)f2bfu(b.y) << 16) | f2bfu(b.x);
    w.w = ((unsigned)f2bfu(b.w) << 16) | f2bfu(b.z);
    *(uint4*)(out + idx / 2) = w;
}

// ===========================================================================
// Kernel 0b: weights fp32 (K x N) -> bf16 transposed (N x K), 4 weights.
// grid (256 /*k*/, 4 /*which weight*/), block 256 (/*n*/).
// ===========================================================================
__global__ __launch_bounds__(256) void convert_wt_kernel(
    const float* __restrict__ Wq, const float* __restrict__ Wk,
    const float* __restrict__ Wv, const float* __restrict__ Wo,
    unsigned short* __restrict__ wt)
{
    int k = blockIdx.x, wsel = blockIdx.y, n = threadIdx.x;
    const float* W = (wsel == 0) ? Wq : (wsel == 1) ? Wk : (wsel == 2) ? Wv : Wo;
    wt[(size_t)wsel * 65536 + (size_t)n * 256 + k] = f2bfu(W[(size_t)k * 256 + n]);
}

// ===========================================================================
// Kernel 1: QKV projection GEMM (bf16 WMMA, async double-buffered staging).
// C = featsbf(8192x256) * W + b.  Block tile 128x64, 8 waves of 2x2 subtiles.
// Q,K stored (B,H,N,DH); V stored transposed (B,H,DH,N) for attention.
// ===========================================================================
__global__ __launch_bounds__(256) void qkv_proj_kernel(
    const unsigned short* __restrict__ abf,   // feats bf16, 8192x256
    const unsigned short* __restrict__ wt,    // Wq^T,Wk^T,Wv^T bf16 concat
    const float* __restrict__ bq, const float* __restrict__ bk,
    const float* __restrict__ bv,
    unsigned short* __restrict__ qws, unsigned short* __restrict__ kws,
    unsigned short* __restrict__ vtws)
{
    constexpr int MT = 128, NT = 64, KT = 32, KDIM = 256, KSTEPS = KDIM / KT;
    __shared__ __attribute__((aligned(32))) unsigned short Alds[2][MT][KT];
    __shared__ __attribute__((aligned(32))) unsigned short Blds[2][NT][KT];

    const int tid  = threadIdx.x;
    const int wave = tid >> 5, lane = tid & 31;
    const int m0 = blockIdx.x * MT, n0 = blockIdx.y * NT;
    const int wsel = blockIdx.z;
    const float* bias = (wsel == 0) ? bq : (wsel == 1) ? bk : bv;
    const unsigned short* wts = wt + (size_t)wsel * 65536;

    const int wr = wave >> 1, wc = wave & 1;
    const int lrow = lane & 15;
    const int akb = (lane < 16) ? 0 : 16;
    const int bkb = (lane < 16) ? 0 : 16;

    v8f acc[2][2] = {};

    auto issue_tiles = [&](int k0, int buf) {
        #pragma unroll
        for (int it = 0; it < 2; ++it) {            // A: 8KB = 512 x 16B
            int idx = tid + it * 256;
            int r = idx >> 2, ch = idx & 3;
            async_copy_b128(&Alds[buf][r][ch * 8],
                            abf + (size_t)(m0 + r) * KDIM + k0 + ch * 8);
        }
        {                                           // B: 4KB = 256 x 16B
            int n = tid >> 2, ch = tid & 3;
            async_copy_b128(&Blds[buf][n][ch * 8],
                            wts + (size_t)(n0 + n) * KDIM + k0 + ch * 8);
        }
    };

    issue_tiles(0, 0);
    for (int t = 0; t < KSTEPS; ++t) {
        const int cur = t & 1;
        if (t + 1 < KSTEPS) {
            __syncthreads();                 // everyone done reading buf cur^1
            issue_tiles((t + 1) * KT, cur ^ 1);
            wait_async<3>();                 // tile t (3 older ops) landed
        } else {
            wait_async<0>();
        }
        __syncthreads();                     // tile t visible to all waves

        v16bf a[2], b[2];
        #pragma unroll
        for (int si = 0; si < 2; ++si) {
            int row = wr * 32 + si * 16 + lrow;
            #pragma unroll
            for (int e = 0; e < 8; ++e) {
                a[si][e]     = u2bf(Alds[cur][row][akb + e]);
                a[si][8 + e] = u2bf(Alds[cur][row][akb + 8 + e]);
            }
        }
        #pragma unroll
        for (int sj = 0; sj < 2; ++sj) {
            int col = wc * 32 + sj * 16 + lrow;
            #pragma unroll
            for (int e = 0; e < 16; ++e)
                b[sj][e] = u2bf(Blds[cur][col][bkb + e]);
        }
        #pragma unroll
        for (int si = 0; si < 2; ++si)
            #pragma unroll
            for (int sj = 0; sj < 2; ++sj)
                acc[si][sj] = wmma_bf16(a[si], b[sj], acc[si][sj]);
    }

    // Epilogue: + bias, store bf16.  Q/K -> (B,H,N,DH); V -> (B,H,DH,N).
    const int rhi = (lane >> 4) * 8;
    #pragma unroll
    for (int si = 0; si < 2; ++si) {
        #pragma unroll
        for (int sj = 0; sj < 2; ++sj) {
            int col = n0 + wc * 32 + sj * 16 + lrow;        // 0..255
            float bval = bias[col];
            int h = col >> 5, d = col & 31;
            #pragma unroll
            for (int i = 0; i < 8; ++i) {
                int row = m0 + wr * 32 + si * 16 + i + rhi; // 0..8191
                int bb = row >> 11, n = row & 2047;
                unsigned short v = f2bfu(acc[si][sj][i] + bval);
                size_t bh = (size_t)bb * 8 + h;
                if (wsel == 2)
                    vtws[(bh * 32 + d) * 2048 + n] = v;     // V^T
                else if (wsel == 1)
                    kws[(bh * 2048 + n) * 32 + d] = v;
                else
                    qws[(bh * 2048 + n) * 32 + d] = v;
            }
        }
    }
}

// ===========================================================================
// Kernel 2: distance-weighted flash attention (per b,h).
// 8 waves x 16 query rows; 32-key tiles async double-buffered into LDS.
// ===========================================================================
__global__ __launch_bounds__(256) void attn_kernel(
    const unsigned short* __restrict__ qws,
    const unsigned short* __restrict__ kws,
    const unsigned short* __restrict__ vtws,
    const float* __restrict__ coords,
    const unsigned char* __restrict__ mask,
    unsigned short* __restrict__ aout)
{
    constexpr int N = 2048, DH = 32, H = 8, KT = 32, NSTEP = N / KT;
    __shared__ __attribute__((aligned(32))) unsigned short Klds[2][KT][DH];
    __shared__ __attribute__((aligned(32))) unsigned short Vtlds[2][DH][KT];
    __shared__ float kclds[2][KT][4];
    __shared__ unsigned char mlds[2][KT];
    __shared__ __attribute__((aligned(32))) unsigned short Plds[8][16][32];

    const int tid = threadIdx.x, wave = tid >> 5, lane = tid & 31;
    const int b = blockIdx.z, h = blockIdx.y;
    const int qbase = blockIdx.x * 128 + wave * 16;
    const int lrow = lane & 15, hi = lane >> 4;

    const unsigned short* qptr  = qws  + (size_t)(b * H + h) * N * DH;
    const unsigned short* kptr  = kws  + (size_t)(b * H + h) * N * DH;
    const unsigned short* vtptr = vtws + (size_t)(b * H + h) * DH * N;

    // Persistent Q A-fragment
    v16bf qa;
    {
        const unsigned short* qrow = qptr + (size_t)(qbase + lrow) * DH;
        const int kb = hi ? 16 : 0;
        #pragma unroll
        for (int e = 0; e < 8; ++e) {
            qa[e]     = u2bf(qrow[kb + e]);
            qa[8 + e] = u2bf(qrow[kb + 8 + e]);
        }
    }
    float qcx[8], qcy[8], qcz[8];
    #pragma unroll
    for (int i = 0; i < 8; ++i) {
        int n = qbase + i + hi * 8;
        qcx[i] = coords[((size_t)b * N + n) * 3 + 0];
        qcy[i] = coords[((size_t)b * N + n) * 3 + 1];
        qcz[i] = coords[((size_t)b * N + n) * 3 + 2];
    }

    v8f o0 = {}, o1 = {};
    float rm[8], rs[8];
    #pragma unroll
    for (int i = 0; i < 8; ++i) { rm[i] = -3.0e38f; rs[i] = 0.0f; }
    const float INV_SQRT_DH = 0.17677669529663687f;

    // Waves 0-3 copy K (2KB), waves 4-7 copy V^T (2KB): 1 async op / thread.
    auto issue_tile = [&](int kb, int buf) {
        if (tid < 128) {
            int r = tid >> 2, ch = tid & 3;
            async_copy_b128(&Klds[buf][r][ch * 8],
                            kptr + (size_t)(kb + r) * DH + ch * 8);
        } else {
            int s = tid - 128, d = s >> 2, ch = s & 3;
            async_copy_b128(&Vtlds[buf][d][ch * 8],
                            vtptr + (size_t)d * N + kb + ch * 8);
        }
        if (tid < 96)
            kclds[buf][tid / 3][tid % 3] =
                coords[((size_t)b * N + kb + tid / 3) * 3 + tid % 3];
        if (tid < 32)
            mlds[buf][tid] = mask[(size_t)b * N + kb + tid];
    };

    issue_tile(0, 0);
    for (int t = 0; t < NSTEP; ++t) {
        const int kb = t * KT, cur = t & 1;
        if (t + 1 < NSTEP) {
            __syncthreads();
            issue_tile(kb + KT, cur ^ 1);
            wait_async<1>();
        } else {
            wait_async<0>();
        }
        __syncthreads();

        // sim = Q K^T (contract over DH=32)
        v16bf bk0, bk1;
        const int dkb = hi ? 16 : 0;
        #pragma unroll
        for (int e = 0; e < 16; ++e) {
            bk0[e] = u2bf(Klds[cur][lrow][dkb + e]);
            bk1[e] = u2bf(Klds[cur][16 + lrow][dkb + e]);
        }
        v8f c0 = {}, c1 = {};
        c0 = wmma_bf16(qa, bk0, c0);
        c1 = wmma_bf16(qa, bk1, c1);

        const float kx0 = kclds[cur][lrow][0],      ky0 = kclds[cur][lrow][1],      kz0 = kclds[cur][lrow][2];
        const float kx1 = kclds[cur][16 + lrow][0], ky1 = kclds[cur][16 + lrow][1], kz1 = kclds[cur][16 + lrow][2];
        const bool mv0 = mlds[cur][lrow] != 0;
        const bool mv1 = mlds[cur][16 + lrow] != 0;

        float l0[8], l1[8];
        #pragma unroll
        for (int i = 0; i < 8; ++i) {
            float dx = qcx[i] - kx0, dy = qcy[i] - ky0, dz = qcz[i] - kz0;
            float w0 = __expf(-(dx * dx + dy * dy + dz * dz) * 0.1f);
            l0[i] = mv0 ? c0[i] * INV_SQRT_DH * w0 : -1.0e9f;
            dx = qcx[i] - kx1; dy = qcy[i] - ky1; dz = qcz[i] - kz1;
            float w1 = __expf(-(dx * dx + dy * dy + dz * dz) * 0.1f);
            l1[i] = mv1 ? c1[i] * INV_SQRT_DH * w1 : -1.0e9f;
        }

        // online softmax: 16-lane butterflies (lanes 0-15 / 16-31 = row halves)
        #pragma unroll
        for (int i = 0; i < 8; ++i) {
            float lm = fmaxf(l0[i], l1[i]);
            #pragma unroll
            for (int m = 1; m < 16; m <<= 1)
                lm = fmaxf(lm, __shfl_xor(lm, m, 32));
            float nm = fmaxf(rm[i], lm);
            float corr = __expf(rm[i] - nm);
            rm[i] = nm;
            float p0 = __expf(l0[i] - nm);
            float p1 = __expf(l1[i] - nm);
            float ps = p0 + p1;
            #pragma unroll
            for (int m = 1; m < 16; m <<= 1)
                ps += __shfl_xor(ps, m, 32);
            rs[i] = rs[i] * corr + ps;
            o0[i] *= corr;
            o1[i] *= corr;
            int r = i + hi * 8;
            Plds[wave][r][lrow]      = f2bfu(p0);
            Plds[wave][r][16 + lrow] = f2bfu(p1);
        }

        // P relayout C->A, then out += P V
        v16bf pa;
        {
            const int cb = hi ? 16 : 0;
            #pragma unroll
            for (int e = 0; e < 8; ++e) {
                pa[e]     = u2bf(Plds[wave][lrow][cb + e]);
                pa[8 + e] = u2bf(Plds[wave][lrow][cb + 8 + e]);
            }
        }
        v16bf bv0, bv1;
        const int vkb = hi ? 16 : 0;
        #pragma unroll
        for (int e = 0; e < 16; ++e) {
            bv0[e] = u2bf(Vtlds[cur][lrow][vkb + e]);
            bv1[e] = u2bf(Vtlds[cur][16 + lrow][vkb + e]);
        }
        o0 = wmma_bf16(pa, bv0, o0);
        o1 = wmma_bf16(pa, bv1, o1);
    }

    #pragma unroll
    for (int i = 0; i < 8; ++i) {
        float inv = 1.0f / rs[i];
        int n = qbase + i + hi * 8;
        size_t base = ((size_t)b * N + n) * 256 + h * 32;
        aout[base + lrow]      = f2bfu(o0[i] * inv);
        aout[base + 16 + lrow] = f2bfu(o1[i] * inv);
    }
}

// ===========================================================================
// Kernel 3: output projection (bf16 WMMA, async staging), fp32 result + bias.
// ===========================================================================
__global__ __launch_bounds__(256) void out_proj_kernel(
    const unsigned short* __restrict__ abf,   // attn out bf16, 8192x256
    const unsigned short* __restrict__ wto,   // Wo^T bf16, 256x256
    const float* __restrict__ bo, float* __restrict__ out)
{
    constexpr int MT = 128, NT = 64, KT = 32, KDIM = 256, KSTEPS = KDIM / KT;
    __shared__ __attribute__((aligned(32))) unsigned short Alds[2][MT][KT];
    __shared__ __attribute__((aligned(32))) unsigned short Blds[2][NT][KT];

    const int tid = threadIdx.x, wave = tid >> 5, lane = tid & 31;
    const int m0 = blockIdx.x * MT, n0 = blockIdx.y * NT;
    const int wr = wave >> 1, wc = wave & 1;
    const int lrow = lane & 15;
    const int akb = (lane < 16) ? 0 : 16;
    const int bkb = (lane < 16) ? 0 : 16;

    v8f acc[2][2] = {};

    auto issue_tiles = [&](int k0, int buf) {
        #pragma unroll
        for (int it = 0; it < 2; ++it) {
            int idx = tid + it * 256;
            int r = idx >> 2, ch = idx & 3;
            async_copy_b128(&Alds[buf][r][ch * 8],
                            abf + (size_t)(m0 + r) * KDIM + k0 + ch * 8);
        }
        {
            int n = tid >> 2, ch = tid & 3;
            async_copy_b128(&Blds[buf][n][ch * 8],
                            wto + (size_t)(n0 + n) * KDIM + k0 + ch * 8);
        }
    };

    issue_tiles(0, 0);
    for (int t = 0; t < KSTEPS; ++t) {
        const int cur = t & 1;
        if (t + 1 < KSTEPS) {
            __syncthreads();
            issue_tiles((t + 1) * KT, cur ^ 1);
            wait_async<3>();
        } else {
            wait_async<0>();
        }
        __syncthreads();

        v16bf a[2], b[2];
        #pragma unroll
        for (int si = 0; si < 2; ++si) {
            int row = wr * 32 + si * 16 + lrow;
            #pragma unroll
            for (int e = 0; e < 8; ++e) {
                a[si][e]     = u2bf(Alds[cur][row][akb + e]);
                a[si][8 + e] = u2bf(Alds[cur][row][akb + 8 + e]);
            }
        }
        #pragma unroll
        for (int sj = 0; sj < 2; ++sj) {
            int col = wc * 32 + sj * 16 + lrow;
            #pragma unroll
            for (int e = 0; e < 16; ++e)
                b[sj][e] = u2bf(Blds[cur][col][bkb + e]);
        }
        #pragma unroll
        for (int si = 0; si < 2; ++si)
            #pragma unroll
            for (int sj = 0; sj < 2; ++sj)
                acc[si][sj] = wmma_bf16(a[si], b[sj], acc[si][sj]);
    }

    const int rhi = (lane >> 4) * 8;
    #pragma unroll
    for (int si = 0; si < 2; ++si) {
        #pragma unroll
        for (int sj = 0; sj < 2; ++sj) {
            int col = n0 + wc * 32 + sj * 16 + lrow;
            float bval = bo[col];
            #pragma unroll
            for (int i = 0; i < 8; ++i) {
                int row = m0 + wr * 32 + si * 16 + i + rhi;
                out[(size_t)row * 256 + col] = acc[si][sj][i] + bval;
            }
        }
    }
}

// ===========================================================================
// Launch
// ===========================================================================
extern "C" void kernel_launch(void* const* d_in, const int* in_sizes, int n_in,
                              void* d_out, int out_size, void* d_ws, size_t ws_size,
                              hipStream_t stream) {
    const float*         feats  = (const float*)d_in[0];
    const float*         coords = (const float*)d_in[1];
    const unsigned char* mask   = (const unsigned char*)d_in[2];
    const float* Wq = (const float*)d_in[3];
    const float* bq = (const float*)d_in[4];
    const float* Wk = (const float*)d_in[5];
    const float* bk = (const float*)d_in[6];
    const float* Wv = (const float*)d_in[7];
    const float* bv = (const float*)d_in[8];
    const float* Wo = (const float*)d_in[9];
    const float* bo = (const float*)d_in[10];
    float* out = (float*)d_out;

    // Workspace layout (bytes)
    char* ws = (char*)d_ws;
    const size_t MiB = 1024 * 1024;
    unsigned short* featsbf = (unsigned short*)(ws);                       // 4 MiB
    unsigned short* wt      = (unsigned short*)(ws + 4 * MiB);             // 512 KiB (4 x 256x256)
    unsigned short* qws     = (unsigned short*)(ws + 4 * MiB + 512 * 1024);
    unsigned short* kws     = (unsigned short*)(ws + 8 * MiB + 512 * 1024);
    unsigned short* vtws    = (unsigned short*)(ws + 12 * MiB + 512 * 1024);
    unsigned short* aws     = (unsigned short*)(ws + 16 * MiB + 512 * 1024);

    // 0) one-time bf16 packing (feats) + transposed bf16 weights
    convert_feats_kernel<<<dim3(1024), 256, 0, stream>>>(feats, (unsigned int*)featsbf);
    convert_wt_kernel<<<dim3(256, 4), 256, 0, stream>>>(Wq, Wk, Wv, Wo, wt);

    // 1) QKV projections (M=8192, N=256, K=256; grid.z selects weight)
    qkv_proj_kernel<<<dim3(64, 4, 3), 256, 0, stream>>>(
        featsbf, wt, bq, bk, bv, qws, kws, vtws);

    // 2) flash attention: grid = (N/128, H, B)
    attn_kernel<<<dim3(16, 8, 4), 256, 0, stream>>>(
        qws, kws, vtws, coords, mask, aws);

    // 3) output projection (Wo^T is 4th transposed weight)
    out_proj_kernel<<<dim3(64, 4, 1), 256, 0, stream>>>(
        aws, wt + (size_t)3 * 65536, bo, out);
}
